// RCNet_36721970381320
// MI455X (gfx1250) — matrix operand order repaired and checked by reading, not compiled, not run on hardware
//
#include <hip/hip_runtime.h>
#include <functional>

typedef _Float16 h16;
typedef __attribute__((ext_vector_type(16))) _Float16 v16h;
typedef __attribute__((ext_vector_type(8)))  _Float16 v8h;
typedef __attribute__((ext_vector_type(8)))  float    v8f;

// ---------------------------------------------------------------------------
// Weight packer: f32 OIHW -> f16 [Cout16][Kpad], K tap-major: k = t*Cpad + ci.
// Grid: (ceil(Kpad/256), Cout16). No runtime division.
// ---------------------------------------------------------------------------
__global__ void k_pack_w(const float* __restrict__ w, h16* __restrict__ wp,
                         int Cout, int CinSrc, int KK, int lCpad, int Kpad) {
  int k = blockIdx.x * blockDim.x + threadIdx.x;
  if (k >= Kpad) return;
  int co = blockIdx.y;
  int t = k >> lCpad;
  int ci = k & ((1 << lCpad) - 1);
  float v = (co < Cout && t < KK && ci < CinSrc)
                ? w[((size_t)co * CinSrc + ci) * KK + t] : 0.f;
  wp[(size_t)co * Kpad + k] = (h16)v;
}

// ---------------------------------------------------------------------------
// Implicit-GEMM conv via V_WMMA_F32_16X16X32_F16, NHWC activations.
//   MODE 0: 3x3 pad1   MODE 1: 1x1   MODE 2: deconv2x (k3,s2,p1,op1)
//   BIG: Cin>=32 -> one uniform tap per 32-chunk, two v8h loads per lane.
// Weight tile [16][Kpad] staged in LDS once per workgroup (4 waves share it).
// Epilogue: +bias, optional *mask, +skip, relu; Y16 NHWC (stride CoutS), Y32 NCHW.
// ---------------------------------------------------------------------------
template <int MODE, bool BIG>
__global__ __launch_bounds__(128) void k_conv_wmma(
    const h16* __restrict__ X, const h16* __restrict__ Wp,
    const float* __restrict__ Bias,
    h16* __restrict__ Y16, float* __restrict__ Y32,
    const h16* __restrict__ Skip, const h16* __restrict__ Mask, int maskBS,
    int lCin, int lHin, int lWin, int Cout, int CoutS, int lHo, int lWo,
    int K, int Kpad, int relu) {
  extern __shared__ h16 lds[];  // [16][Kpad] weight tile
  const int lane = threadIdx.x;
  const int n = lane & 15;
  const int g = lane >> 4;
  const int co0 = blockIdx.y * 16;
  const int b = blockIdx.z;
  const int ptile = blockIdx.x * blockDim.y + threadIdx.y;
  const int p = ptile * 16 + n;
  const int Wo = 1 << lWo;
  const int oy = p >> lWo, ox = p & (Wo - 1);
  const int Hin = 1 << lHin, Win = 1 << lWin;
  const int Cin = 1 << lCin;

  // ---- cooperative weight-tile stage: 16 contiguous rows of Wp -> LDS -----
  {
    const h16* src = Wp + (size_t)co0 * Kpad;
    const int tid = threadIdx.y * 32 + lane;
    const int tot = 16 * Kpad;  // multiple of 1024 halves? (mult of 512, ok)
    for (int ofs = tid * 8; ofs < tot; ofs += 128 * 8)
      *(v8h*)(lds + ofs) = *(const v8h*)(src + ofs);
  }
  __syncthreads();

  const h16* wrow = lds + (size_t)n * Kpad;  // A row M = lane&15

  v8f acc0, acc1;
#pragma unroll
  for (int i = 0; i < 8; ++i) { acc0[i] = 0.f; acc1[i] = 0.f; }

  auto chunk = [&](int kc, v8f& accr) {
    // A fragment (contiguous layout matching B: k = kc + 16*g + h)
    v8h a0 = *(const v8h*)(wrow + kc + 16 * g);
    v8h a1 = *(const v8h*)(wrow + kc + 16 * g + 8);
    v16h A;
#pragma unroll
    for (int i = 0; i < 8; ++i) { A[i] = a0[i]; A[i + 8] = a1[i]; }

    // B fragment (contiguous layout matching A: k = kc + 16*g + h, column = n)
    v16h Bm;
    if constexpr (BIG) {  // whole chunk within one tap
      int iy, ix; bool ok = true;
      if constexpr (MODE == 1) { iy = oy; ix = ox; }
      else {
        int t = kc >> lCin;
        int r = t / 3, s = t - r * 3;
        if constexpr (MODE == 0) {
          iy = oy + r - 1; ix = ox + s - 1;
          ok = (iy >= 0) && (iy < Hin) && (ix >= 0) && (ix < Win);
        } else {
          int qy = oy + r - 1, qx = ox + s - 1;
          ok = (qy >= 0) && (qx >= 0) && !(qy & 1) && !(qx & 1);
          iy = qy >> 1; ix = qx >> 1;
          ok = ok && (iy < Hin) && (ix < Win);
        }
      }
      int ci0 = (kc & (Cin - 1)) + 16 * g;
      v8h b0, b1;
#pragma unroll
      for (int i = 0; i < 8; ++i) { b0[i] = (h16)0.f; b1[i] = (h16)0.f; }
      if (ok) {
        const h16* src = X +
            ((((((size_t)b << lHin) + iy) << lWin) + ix) << lCin) + ci0;
        b0 = *(const v8h*)src;
        b1 = *(const v8h*)(src + 8);
      }
#pragma unroll
      for (int i = 0; i < 8; ++i) { Bm[i] = b0[i]; Bm[i + 8] = b1[i]; }
    } else {  // small-channel fallback (only MODE 0): per-element gather
#pragma unroll
      for (int h = 0; h < 16; ++h) {
        int k = kc + 16 * g + h;
        h16 v = (h16)0.f;
        if (k < K) {
          int t = k >> lCin, ci = k & (Cin - 1);
          int r = t / 3, s = t - r * 3;
          int iy = oy + r - 1, ix = ox + s - 1;
          if ((iy >= 0) && (iy < Hin) && (ix >= 0) && (ix < Win))
            v = X[((((((size_t)b << lHin) + iy) << lWin) + ix) << lCin) + ci];
        }
        Bm[h] = v;
      }
    }
    accr = __builtin_amdgcn_wmma_f32_16x16x32_f16(false, A, false, Bm,
                                                  (short)0, accr, false, false);
  };

  int kc = 0;
  for (; kc + 64 <= K; kc += 64) { chunk(kc, acc0); chunk(kc + 32, acc1); }
  for (; kc < K; kc += 32) chunk(kc, acc0);
#pragma unroll
  for (int i = 0; i < 8; ++i) acc0[i] += acc1[i];

  float mval = 1.f;
  if (Mask) mval = (float)Mask[(size_t)b * maskBS + p];
  const size_t pixBase = ((size_t)b << (lHo + lWo)) + p;
#pragma unroll
  for (int r = 0; r < 8; ++r) {
    int co = co0 + r + 8 * g;  // C/D layout: VGPR r -> M = r + 8*(lane/16)
    if (co >= Cout) continue;
    float v = acc0[r] + Bias[co];
    if (Mask) v *= mval;
    size_t o16 = pixBase * CoutS + co;
    if (Skip) v += (float)Skip[o16];
    if (relu) v = fmaxf(v, 0.f);
    if (Y16) Y16[o16] = (h16)v;
    if (Y32) Y32[(((size_t)b * Cout + co) << (lHo + lWo)) + p] = v;
  }
}

// ---------------------------------------------------------------------------
// Grouped (68 groups, 1ch) 3x3 conv, NHWC [B][128][128][68]. VALU path (K=9).
// ---------------------------------------------------------------------------
__global__ void k_lmconv(const h16* __restrict__ X, const float* __restrict__ W,
                         const float* __restrict__ Bi, h16* __restrict__ Y,
                         int total, int layer) {
  int i = blockIdx.x * blockDim.x + threadIdx.x;
  if (i >= total) return;
  int c = i % 68;
  int rest = i / 68;
  int x = rest & 127;
  int y = (rest >> 7) & 127;
  int b = rest >> 14;
  const float* wk = W + ((size_t)layer * 68 + c) * 9;
  float acc = Bi[layer * 68 + c];
  for (int r = 0; r < 3; ++r)
    for (int s = 0; s < 3; ++s) {
      int iy = y + r - 1, ix = x + s - 1;
      if (iy >= 0 && iy < 128 && ix >= 0 && ix < 128)
        acc += wk[r * 3 + s] *
               (float)X[(((((size_t)b << 7) | iy) << 7) | ix) * 68 + c];
    }
  Y[i] = (h16)acc;
}

// masks from NHWC [B][128][128][68] -> planes [B][4][16384]
__global__ void k_masks(const h16* __restrict__ LM, h16* __restrict__ M,
                        int total) {
  int i = blockIdx.x * blockDim.x + threadIdx.x;
  if (i >= total) return;
  int p = i & 16383, b = i >> 14;
  const h16* base = LM + (((size_t)b << 14) | p) * 68;
  float eye = 0, nose = 0, jaw = 0, mouth = 0;
  for (int c = 17; c < 27; ++c) eye += (float)base[c];
  for (int c = 36; c < 48; ++c) eye += (float)base[c];
  for (int c = 27; c < 36; ++c) nose += (float)base[c];
  for (int c = 0; c < 17; ++c) jaw += (float)base[c];
  for (int c = 48; c < 68; ++c) mouth += (float)base[c];
  size_t ob = ((size_t)b << 16) | p;  // b*4*16384 + p
  M[ob] = (h16)eye; M[ob + 16384] = (h16)nose;
  M[ob + 2 * 16384] = (h16)jaw; M[ob + 3 * 16384] = (h16)mouth;
}

// NHWC maxpool (C = 128)
__global__ void k_maxpool(const h16* __restrict__ X, h16* __restrict__ Y,
                          int total, int lC, int lHin, int lWin, int lHo,
                          int lWo, int kk, int ss, int pad) {
  int i = blockIdx.x * blockDim.x + threadIdx.x;
  if (i >= total) return;
  int c = i & ((1 << lC) - 1);
  int x = (i >> lC) & ((1 << lWo) - 1);
  int y = (i >> (lC + lWo)) & ((1 << lHo) - 1);
  int b = i >> (lC + lWo + lHo);
  int Hin = 1 << lHin, Win = 1 << lWin;
  float m = -3.4e38f;
  for (int r = 0; r < kk; ++r)
    for (int s = 0; s < kk; ++s) {
      int iy = y * ss - pad + r, ix = x * ss - pad + s;
      if (iy >= 0 && iy < Hin && ix >= 0 && ix < Win)
        m = fmaxf(m, (float)X[((((((size_t)b << lHin) + iy) << lWin) + ix)
                               << lC) + c]);
    }
  Y[i] = (h16)m;
}

// out = up1 + nearest-up2x(low), NHWC
__global__ void k_upadd(const h16* __restrict__ Up, const h16* __restrict__ Low,
                        h16* __restrict__ Y, int total, int lC, int lH, int lW) {
  int i = blockIdx.x * blockDim.x + threadIdx.x;
  if (i >= total) return;
  int c = i & ((1 << lC) - 1);
  int x = (i >> lC) & ((1 << lW) - 1);
  int y = (i >> (lC + lW)) & ((1 << lH) - 1);
  int b = i >> (lC + lW + lH);
  size_t lo = ((((((size_t)b << (lH - 1)) + (y >> 1)) << (lW - 1)) + (x >> 1))
               << lC) + c;
  Y[i] = (h16)((float)Up[i] + (float)Low[lo]);
}

// pixel_shuffle r=2, NHWC: X [B][Ho/2][Wo/2][Co*4] -> Y [B][Ho][Wo][Co]
__global__ void k_pshuf(const h16* __restrict__ X, h16* __restrict__ Y,
                        int total, int lCo, int lHo, int lWo) {
  int i = blockIdx.x * blockDim.x + threadIdx.x;
  if (i >= total) return;
  int c = i & ((1 << lCo) - 1);
  int x = (i >> lCo) & ((1 << lWo) - 1);
  int y = (i >> (lCo + lWo)) & ((1 << lHo) - 1);
  int b = i >> (lCo + lWo + lHo);
  int ci = (c << 2) + ((y & 1) << 1) + (x & 1);
  Y[i] = X[((((((size_t)b << (lHo - 1)) + (y >> 1)) << (lWo - 1)) + (x >> 1))
            << (lCo + 2)) + ci];
}

// f = s1[::8] + 3*s2[::4] + 3*s3[::2] + s4, NHWC, C=128, out 16x16
__global__ void k_combine(const h16* __restrict__ s1, const h16* __restrict__ s2,
                          const h16* __restrict__ s3, const h16* __restrict__ s4,
                          h16* __restrict__ F, int total) {
  int i = blockIdx.x * blockDim.x + threadIdx.x;
  if (i >= total) return;
  int c = i & 127;
  int x = (i >> 7) & 15;
  int y = (i >> 11) & 15;
  int b = i >> 15;
  float v = (float)s1[((((((size_t)b << 7) + (y << 3)) << 7) + (x << 3)) << 7) + c] +
      3.f * (float)s2[((((((size_t)b << 6) + (y << 2)) << 6) + (x << 2)) << 7) + c] +
      3.f * (float)s3[((((((size_t)b << 5) + (y << 1)) << 5) + (x << 1)) << 7) + c] +
            (float)s4[((((((size_t)b << 4) + y) << 4) + x) << 7) + c];
  F[i] = (h16)v;
}

// NCHW f32 [B][3][16384] -> NHWC f16 [B][16384][4] (channel 3 untouched)
__global__ void k_padcvt(const float* __restrict__ X, h16* __restrict__ Y,
                         int total) {
  int i = blockIdx.x * blockDim.x + threadIdx.x;
  if (i >= total) return;
  int p = i & 16383;
  int bc = i >> 14;
  int c = bc % 3, b = bc / 3;
  Y[((((size_t)b << 14) | p) << 2) + c] = (h16)X[i];
}

__global__ void k_zero16(h16* __restrict__ Y, size_t nn) {
  size_t i = (size_t)blockIdx.x * blockDim.x + threadIdx.x;
  if (i < nn) Y[i] = (h16)0.f;
}

// ---------------------------------------------------------------------------
// Host orchestration
// ---------------------------------------------------------------------------
extern "C" void kernel_launch(void* const* d_in, const int* in_sizes, int n_in,
                              void* d_out, int out_size, void* d_ws, size_t ws_size,
                              hipStream_t stream) {
  (void)in_sizes; (void)n_in; (void)out_size; (void)ws_size;
  const int Bn = 4;
  const float* x32 = (const float*)d_in[0];
  int cur = 1;
  auto nf = [&]() { return (const float*)d_in[cur++]; };
  auto lg = [](int v) { return 31 - __builtin_clz((unsigned)v); };

  size_t off = 0;
  auto allocB = [&](size_t bytes) {
    void* p = (char*)d_ws + off;
    off += (bytes + 63) & ~(size_t)63;
    return p;
  };
  auto A16 = [&](size_t n) { return (h16*)allocB(n * sizeof(h16)); };
  auto NB = [](size_t n) { return (unsigned)((n + 255) / 256); };

  struct PW { const h16* wp; const float* b; int Cout, lC, K, Kpad; };
  auto packConv = [&](int Cout, int CinSrc, int kh, int Cpad) -> PW {
    const float* w = nf();
    const float* bb = nf();
    int KK = kh * kh;
    int K = KK * Cpad;
    int Kpad = (K + 31) & ~31;
    int C16 = (Cout + 15) & ~15;
    h16* wp = (h16*)allocB((size_t)C16 * Kpad * sizeof(h16));
    dim3 grd(NB(Kpad), (unsigned)C16);
    k_pack_w<<<grd, 256, 0, stream>>>(w, wp, Cout, CinSrc, KK, lg(Cpad), Kpad);
    return PW{wp, bb, Cout, lg(Cpad), K, Kpad};
  };
  auto pc = [&](int Cout, int Cin, int kh) { return packConv(Cout, Cin, kh, Cin); };

  // ---- pack all weights in setup_inputs() insertion order -----------------
  PW sh_c1 = packConv(128, 3, 3, 4);
  PW r1c1 = pc(128, 128, 3), r1c2 = pc(128, 128, 3);
  PW r2[2][2];
  for (int i = 0; i < 2; ++i) { r2[i][0] = pc(128, 128, 3); r2[i][1] = pc(128, 128, 3); }
  PW r3[3][2];
  for (int i = 0; i < 3; ++i) { r3[i][0] = pc(128, 128, 3); r3[i][1] = pc(128, 128, 3); }

  PW dec[3];
  for (int i = 0; i < 3; ++i) dec[i] = pc(128, 128, 3);

  struct RSD { PW c1, c2, c3; };
  auto packRSD = [&]() -> RSD {
    RSD r;
    r.c1 = pc(64, 128, 1);
    r.c2 = pc(64, 64, 3);
    r.c3 = pc(128, 64, 1);
    return r;
  };
  RSD hg_up[4], hg_low1[4], hg_low3[4], hg_low2;
  std::function<void(int)> packHG = [&](int nlev) {
    int d = 4 - nlev;
    hg_up[d] = packRSD();
    hg_low1[d] = packRSD();
    if (nlev > 1) packHG(nlev - 1); else hg_low2 = packRSD();
    hg_low3[d] = packRSD();
  };
  packHG(4);
  PW ld_conv = pc(68, 128, 3);

  PW u_c0 = pc(128, 128, 3);
  PW u_c1 = pc(64, 32, 3);
  PW r1a0 = pc(64, 64, 3), r1a1 = pc(64, 64, 3);
  PW r1b0 = pc(64, 64, 3), r1b1 = pc(64, 64, 3);
  PW u_c2 = pc(32, 16, 3);
  PW r2a0 = pc(32, 32, 3), r2a1 = pc(32, 32, 3);
  PW r2b0 = pc(32, 32, 3), r2b1 = pc(32, 32, 3);
  PW u_c3 = pc(16, 8, 3);
  PW r3a0 = pc(16, 16, 3), r3a1 = pc(16, 16, 3);
  PW r3b0 = pc(16, 16, 3), r3b1 = pc(16, 16, 3);

  PW gc = pc(3, 16, 3);
  PW gr = pc(3, 16, 3);
  PW fea[4][3];
  for (int t = 0; t < 4; ++t)
    for (int l = 0; l < 3; ++l) fea[t][l] = pc(16, 16, 3);
  const float* lm_w = nf();
  const float* lm_b = nf();

  // ---- NHWC f16 activation buffers (L2-resident working set) --------------
  const int NP128 = 128 * 128;
  h16* a_x  = A16((size_t)Bn * NP128 * 4);
  h16* s1   = A16((size_t)Bn * NP128 * 128);
  h16* h1   = A16((size_t)Bn * 64 * 64 * 128);
  h16* h2   = A16((size_t)Bn * 64 * 64 * 128);
  h16* q1   = A16((size_t)Bn * 32 * 32 * 128);
  h16* q2   = A16((size_t)Bn * 32 * 32 * 128);
  h16* e1   = A16((size_t)Bn * 16 * 16 * 128);
  h16* e2   = A16((size_t)Bn * 16 * 16 * 128);
  h16* fB   = A16((size_t)Bn * 16 * 16 * 128);
  h16* d32  = A16((size_t)Bn * 32 * 32 * 128);
  h16* d64  = A16((size_t)Bn * 64 * 64 * 128);
  h16* d128 = A16((size_t)Bn * NP128 * 128);
  h16* t64a = A16((size_t)Bn * NP128 * 64);
  h16* t64b = A16((size_t)Bn * NP128 * 64);
  h16* hup0 = A16((size_t)Bn * NP128 * 128);
  h16* hx1  = A16((size_t)Bn * 64 * 64 * 128);
  h16* hup1 = A16((size_t)Bn * 64 * 64 * 128);
  h16* hx2  = A16((size_t)Bn * 32 * 32 * 128);
  h16* hup2 = A16((size_t)Bn * 32 * 32 * 128);
  h16* hx3  = A16((size_t)Bn * 16 * 16 * 128);
  h16* hup3 = A16((size_t)Bn * 16 * 16 * 128);
  h16* hx4  = A16((size_t)Bn * 8 * 8 * 128);
  h16* heat = A16((size_t)Bn * NP128 * 68);
  h16* u0   = A16((size_t)Bn * 16 * 16 * 128);
  h16* ps1  = A16((size_t)Bn * 32 * 32 * 32);
  h16* ua32 = A16((size_t)Bn * 32 * 32 * 64);
  h16* ub32 = A16((size_t)Bn * 32 * 32 * 64);
  h16* ps2  = A16((size_t)Bn * 64 * 64 * 16);
  h16* ua64 = A16((size_t)Bn * 64 * 64 * 32);
  h16* ub64 = A16((size_t)Bn * 64 * 64 * 32);
  h16* ps3  = A16((size_t)Bn * NP128 * 8);
  h16* feats = A16((size_t)Bn * NP128 * 16);
  h16* tw   = A16((size_t)Bn * NP128 * 16);
  h16* twb  = A16((size_t)Bn * NP128 * 16);
  h16* lmA  = A16((size_t)Bn * NP128 * 68);
  h16* lmB  = A16((size_t)Bn * NP128 * 68);
  h16* mk   = A16((size_t)Bn * 4 * NP128);
  h16* fused = A16((size_t)Bn * NP128 * 16);
  h16* refined = A16((size_t)Bn * NP128 * 4);

  auto convL = [&](const h16* X, const PW& pw, h16* Y16, float* Y32,
                   const h16* Skip, const h16* Mask, int maskBS,
                   int Hin, int Win, int Ho, int Wo, int CoutS,
                   int mode, int relu) {
    int C16 = (pw.Cout + 15) & ~15;
    dim3 blk(32, 4, 1);
    dim3 grd((unsigned)(Ho * Wo / 64), (unsigned)(C16 / 16), (unsigned)Bn);
    size_t shmem = (size_t)16 * pw.Kpad * sizeof(h16);
    if (mode == 0) {
      if (pw.lC >= 5)
        k_conv_wmma<0, true><<<grd, blk, shmem, stream>>>(
            X, pw.wp, pw.b, Y16, Y32, Skip, Mask, maskBS, pw.lC, lg(Hin),
            lg(Win), pw.Cout, CoutS, lg(Ho), lg(Wo), pw.K, pw.Kpad, relu);
      else
        k_conv_wmma<0, false><<<grd, blk, shmem, stream>>>(
            X, pw.wp, pw.b, Y16, Y32, Skip, Mask, maskBS, pw.lC, lg(Hin),
            lg(Win), pw.Cout, CoutS, lg(Ho), lg(Wo), pw.K, pw.Kpad, relu);
    } else if (mode == 1) {
      k_conv_wmma<1, true><<<grd, blk, shmem, stream>>>(
          X, pw.wp, pw.b, Y16, Y32, Skip, Mask, maskBS, pw.lC, lg(Hin),
          lg(Win), pw.Cout, CoutS, lg(Ho), lg(Wo), pw.K, pw.Kpad, relu);
    } else {
      k_conv_wmma<2, true><<<grd, blk, shmem, stream>>>(
          X, pw.wp, pw.b, Y16, Y32, Skip, Mask, maskBS, pw.lC, lg(Hin),
          lg(Win), pw.Cout, CoutS, lg(Ho), lg(Wo), pw.K, pw.Kpad, relu);
    }
  };
  // resblock: x = relu(x + conv(relu(conv(x)))) -- in-place on x
  auto rbL = [&](const PW& c1, const PW& c2, h16* x, h16* tmp, int R, int C) {
    convL(x, c1, tmp, nullptr, nullptr, nullptr, 0, R, R, R, R, C, 0, 1);
    convL(tmp, c2, x, nullptr, x, nullptr, 0, R, R, R, R, C, 0, 1);
  };
  // bottleneck residual: out = xin + c3(relu(c2(relu(c1(xin)))))
  auto resid = [&](const RSD& rp, const h16* xin, h16* out, int R) {
    convL(xin, rp.c1, t64a, nullptr, nullptr, nullptr, 0, R, R, R, R, 64, 1, 1);
    convL(t64a, rp.c2, t64b, nullptr, nullptr, nullptr, 0, R, R, R, R, 64, 0, 1);
    convL(t64b, rp.c3, out, nullptr, xin, nullptr, 0, R, R, R, R, 128, 1, 0);
  };

  float* outF = (float*)d_out;
  const size_t CO_STRIDE = (size_t)Bn * 3 * NP128;     // 196608
  const size_t REF_BASE = 4 * CO_STRIDE;               // 786432
  const size_t HEAT_BASE = 8 * CO_STRIDE;              // 1572864
  const size_t HEAT_STRIDE = (size_t)Bn * 68 * NP128;  // 4456448

  // zero-init 4-channel padded input & refined (channel 3 stays zero)
  k_zero16<<<NB((size_t)Bn * NP128 * 4), 256, 0, stream>>>(a_x, (size_t)Bn * NP128 * 4);
  k_zero16<<<NB((size_t)Bn * NP128 * 4), 256, 0, stream>>>(refined, (size_t)Bn * NP128 * 4);
  k_padcvt<<<NB((size_t)Bn * 3 * NP128), 256, 0, stream>>>(x32, a_x, Bn * 3 * NP128);

  for (int step = 0; step < 4; ++step) {
    const h16* xin = (step == 0) ? a_x : refined;

    // ---- shared_feature ----
    convL(xin, sh_c1, s1, nullptr, nullptr, nullptr, 0, 128, 128, 128, 128, 128, 0, 0);
    k_maxpool<<<NB((size_t)Bn * 64 * 64 * 128), 256, 0, stream>>>(
        s1, h1, Bn * 64 * 64 * 128, 7, 7, 7, 6, 6, 3, 2, 1);
    rbL(r1c1, r1c2, h1, h2, 64, 128);                  // s2 = h1
    k_maxpool<<<NB((size_t)Bn * 32 * 32 * 128), 256, 0, stream>>>(
        h1, q1, Bn * 32 * 32 * 128, 7, 6, 6, 5, 5, 3, 2, 1);
    for (int i = 0; i < 2; ++i) rbL(r2[i][0], r2[i][1], q1, q2, 32, 128);  // s3
    k_maxpool<<<NB((size_t)Bn * 16 * 16 * 128), 256, 0, stream>>>(
        q1, e1, Bn * 16 * 16 * 128, 7, 5, 5, 4, 4, 3, 2, 1);
    for (int i = 0; i < 3; ++i) rbL(r3[i][0], r3[i][1], e1, e2, 16, 128);  // s4
    k_combine<<<NB((size_t)Bn * 256 * 128), 256, 0, stream>>>(
        s1, h1, q1, e1, fB, Bn * 256 * 128);

    // ---- landmark_es: deconvs + hourglass + heat ----
    convL(fB, dec[0], d32, nullptr, nullptr, nullptr, 0, 16, 16, 32, 32, 128, 2, 1);
    convL(d32, dec[1], d64, nullptr, nullptr, nullptr, 0, 32, 32, 64, 64, 128, 2, 1);
    convL(d64, dec[2], d128, nullptr, nullptr, nullptr, 0, 64, 64, 128, 128, 128, 2, 1);

    h16* hx[5] = {d128, hx1, hx2, hx3, hx4};
    h16* hup[4] = {hup0, hup1, hup2, hup3};
    const int res[5] = {128, 64, 32, 16, 8};
    for (int d = 0; d < 4; ++d) {
      resid(hg_up[d], hx[d], hup[d], res[d]);
      int Rl = res[d + 1];
      k_maxpool<<<NB((size_t)Bn * Rl * Rl * 128), 256, 0, stream>>>(
          hx[d], hx[d + 1], Bn * Rl * Rl * 128, 7, lg(res[d]), lg(res[d]),
          lg(Rl), lg(Rl), 2, 2, 0);
      resid(hg_low1[d], hx[d + 1], hx[d + 1], Rl);
    }
    resid(hg_low2, hx[4], hx[4], 8);
    for (int d = 3; d >= 0; --d) {
      resid(hg_low3[d], hx[d + 1], hx[d + 1], res[d + 1]);
      k_upadd<<<NB((size_t)Bn * res[d] * res[d] * 128), 256, 0, stream>>>(
          hup[d], hx[d + 1], hx[d], Bn * res[d] * res[d] * 128, 7,
          lg(res[d]), lg(res[d]));
    }
    convL(hx[0], ld_conv, heat, outF + HEAT_BASE + (size_t)step * HEAT_STRIDE,
          nullptr, nullptr, 0, 128, 128, 128, 128, 68, 0, 1);

    // ---- ups ----
    convL(fB, u_c0, u0, nullptr, nullptr, nullptr, 0, 16, 16, 16, 16, 128, 0, 1);
    k_pshuf<<<NB((size_t)Bn * 32 * 32 * 32), 256, 0, stream>>>(
        u0, ps1, Bn * 32 * 32 * 32, 5, 5, 5);
    convL(ps1, u_c1, ua32, nullptr, nullptr, nullptr, 0, 32, 32, 32, 32, 64, 0, 0);
    rbL(r1a0, r1a1, ua32, ub32, 32, 64);
    rbL(r1b0, r1b1, ua32, ub32, 32, 64);
    k_pshuf<<<NB((size_t)Bn * 64 * 64 * 16), 256, 0, stream>>>(
        ua32, ps2, Bn * 64 * 64 * 16, 4, 6, 6);
    convL(ps2, u_c2, ua64, nullptr, nullptr, nullptr, 0, 64, 64, 64, 64, 32, 0, 0);
    rbL(r2a0, r2a1, ua64, ub64, 64, 32);
    rbL(r2b0, r2b1, ua64, ub64, 64, 32);
    k_pshuf<<<NB((size_t)Bn * NP128 * 8), 256, 0, stream>>>(
        ua64, ps3, Bn * NP128 * 8, 3, 7, 7);
    convL(ps3, u_c3, feats, nullptr, nullptr, nullptr, 0, 128, 128, 128, 128, 16, 0, 0);
    rbL(r3a0, r3a1, feats, tw, 128, 16);
    rbL(r3b0, r3b1, feats, tw, 128, 16);

    // ---- getcoarse (f32 NCHW only) ----
    convL(feats, gc, nullptr, outF + (size_t)step * CO_STRIDE,
          nullptr, nullptr, 0, 128, 128, 128, 128, 4, 0, 1);

    // ---- component_awa ----
    const h16* lmin = heat;
    h16* lps[2] = {lmA, lmB};
    for (int l = 0; l < 5; ++l) {
      h16* o = lps[l & 1];
      k_lmconv<<<NB((size_t)Bn * NP128 * 68), 256, 0, stream>>>(
          lmin, lm_w, lm_b, o, Bn * NP128 * 68, l);
      lmin = o;
    }
    k_masks<<<NB((size_t)Bn * NP128), 256, 0, stream>>>(lmin, mk, Bn * NP128);
    k_zero16<<<NB((size_t)Bn * NP128 * 16), 256, 0, stream>>>(
        fused, (size_t)Bn * NP128 * 16);
    for (int t = 0; t < 4; ++t) {
      convL(feats, fea[t][0], tw, nullptr, nullptr, nullptr, 0, 128, 128, 128, 128, 16, 0, 0);
      convL(tw, fea[t][1], twb, nullptr, nullptr, nullptr, 0, 128, 128, 128, 128, 16, 0, 0);
      convL(twb, fea[t][2], fused, nullptr, fused, mk + (size_t)t * NP128,
            4 * NP128, 128, 128, 128, 128, 16, 0, 0);
    }

    // ---- getrefined (f16 NHWC padded 4ch + f32 NCHW out) ----
    convL(fused, gr, refined, outF + REF_BASE + (size_t)step * CO_STRIDE,
          nullptr, nullptr, 0, 128, 128, 128, 128, 4, 0, 1);
  }
}
